// CausalSelfAttention_17136919511181
// MI455X (gfx1250) — matrix operand order repaired and checked by reading, not compiled, not run on hardware
//
#include <hip/hip_runtime.h>

// ---------------------------------------------------------------------------
// CDNA5 (gfx1250) causal self-attention, f16 WMMA with f32 accumulate.
// B=4, T=2048, C=1024, H=16, HD=64. Mask: tril AND (j % 25 != 24).
// ---------------------------------------------------------------------------

typedef __attribute__((ext_vector_type(16))) _Float16 v16h;
typedef __attribute__((ext_vector_type(8)))  _Float16 v8h;
typedef __attribute__((ext_vector_type(4)))  _Float16 v4h;
typedef __attribute__((ext_vector_type(8)))  float    v8f;
typedef __attribute__((ext_vector_type(4)))  float    v4f;

#define BB   4
#define TT   2048
#define CC   1024
#define HH   16
#define HD   64
#define MTOT (BB * TT)          // 8192 rows for the projections

// ---- WMMA wrapper ----------------------------------------------------------
__device__ __forceinline__ v8f wmma_f16(v16h a, v16h b, v8f c) {
  // 8 args: (neg_a, A, neg_b, B, c_mod, C, reuse_a, reuse_b)
  return __builtin_amdgcn_wmma_f32_16x16x32_f16(false, a, false, b,
                                                (short)0, c, false, false);
}

// ---- Operand loaders (ISA 16-bit A/B VGPR layouts, wave32) -----------------
// A 16x32: lane L = row L%16; halves at k = (L/16)*8 + 0..7 and +16.
// Caller passes p already offset by row*ld + (L>>4)*8 + k0.
__device__ __forceinline__ v16h ld_a(const _Float16* p) {
  v8h lo = *(const v8h*)p;
  v8h hi = *(const v8h*)(p + 16);
  return __builtin_shufflevector(lo, hi, 0, 1, 2, 3, 4, 5, 6, 7,
                                 8, 9, 10, 11, 12, 13, 14, 15);
}
// B 32x16: lane L = col L%16; halves at k = (L/16)*8 + 0..7 and +16.
// Caller passes p already offset by col*ld + (L>>4)*8 + k0.
__device__ __forceinline__ v16h ld_b(const _Float16* p) {
  v8h lo = *(const v8h*)p;
  v8h hi = *(const v8h*)(p + 16);
  return __builtin_shufflevector(lo, hi, 0, 1, 2, 3, 4, 5, 6, 7,
                                 8, 9, 10, 11, 12, 13, 14, 15);
}

// ---- fp32 -> f16 conversion (vectorized: b128 in, b64 out) -----------------
__global__ void cvt_f32_f16(const float* __restrict__ in,
                            _Float16* __restrict__ out, int n4) {
  const v4f* __restrict__ in4 = (const v4f*)in;
  v4h* __restrict__ out4 = (v4h*)out;
  int i = blockIdx.x * blockDim.x + threadIdx.x;
  int stride = gridDim.x * blockDim.x;
  for (; i < n4; i += stride) {
    v4f v = in4[i];
    v4h o;
#pragma unroll
    for (int j = 0; j < 4; ++j) o[j] = (_Float16)v[j];
    out4[i] = o;
  }
}

// ---- WMMA GEMM: out[m][n] = sum_k A[m][k] * W[n][k] + bias[n] --------------
// A: [M][1024] f16 row-major. W: [1024][1024] f16 row-major (torch layout).
// mode 0: f16 out [M][1024]; mode 1: f16 out transposed per head [B,H,HD,T];
// mode 2: f32 out [M][1024].
// One wave computes a 32(M) x 64(N) tile = 8 WMMA accumulators. The k-loop is
// unrolled by 64 with explicit ping-pong operand buffers: compute from buffer
// A while buffer B's 12 b128 loads are in flight (and vice versa). No buffer
// rotation copies; s_wait_loadcnt waits are partial.
__global__ void __launch_bounds__(128)
gemm_wmma(const _Float16* __restrict__ A, const _Float16* __restrict__ W,
          const float* __restrict__ bias, void* __restrict__ outp, int mode) {
  const int lane    = threadIdx.x & 31;
  const int wave    = blockIdx.x * (blockDim.x >> 5) + (threadIdx.x >> 5);
  const int nGroups = CC / 64;                    // 16
  const int mBase   = (wave / nGroups) * 32;
  const int nBase   = (wave % nGroups) * 64;
  if (mBase >= MTOT) return;                      // wave-uniform

  const int colLane = lane & 15;
  const int kA = (lane >> 4) << 3;                // A k-offset per lane half
  const int kB = (lane >> 4) << 3;                // B k-offset per lane half

  const _Float16* Ar0 = A + (size_t)(mBase + colLane) * CC + kA;
  const _Float16* Ar1 = Ar0 + (size_t)16 * CC;
  const _Float16* Wr[4];
#pragma unroll
  for (int t = 0; t < 4; ++t)
    Wr[t] = W + (size_t)(nBase + t * 16 + colLane) * CC + kB;

  v8f acc[8] = {};                                // acc[t] (m0), acc[4+t] (m1)

  // ping-pong operand buffers
  v16h a0A, a1A, bA[4], a0B, a1B, bB[4];

  // prologue: buffer A <- k = 0
  a0A = ld_a(Ar0); a1A = ld_a(Ar1);
#pragma unroll
  for (int t = 0; t < 4; ++t) bA[t] = ld_b(Wr[t]);

  int k0 = 0;
  for (; k0 < CC - 64; k0 += 64) {
    // buffer B <- k0 + 32
    a0B = ld_a(Ar0 + k0 + 32); a1B = ld_a(Ar1 + k0 + 32);
#pragma unroll
    for (int t = 0; t < 4; ++t) bB[t] = ld_b(Wr[t] + k0 + 32);
    // compute buffer A (k0)
#pragma unroll
    for (int t = 0; t < 4; ++t) {
      acc[t]     = wmma_f16(a0A, bA[t], acc[t]);
      acc[4 + t] = wmma_f16(a1A, bA[t], acc[4 + t]);
    }
    // buffer A <- k0 + 64
    a0A = ld_a(Ar0 + k0 + 64); a1A = ld_a(Ar1 + k0 + 64);
#pragma unroll
    for (int t = 0; t < 4; ++t) bA[t] = ld_b(Wr[t] + k0 + 64);
    // compute buffer B (k0 + 32)
#pragma unroll
    for (int t = 0; t < 4; ++t) {
      acc[t]     = wmma_f16(a0B, bB[t], acc[t]);
      acc[4 + t] = wmma_f16(a1B, bB[t], acc[4 + t]);
    }
  }
  // peeled tail: k0 == CC - 64
  a0B = ld_a(Ar0 + k0 + 32); a1B = ld_a(Ar1 + k0 + 32);
#pragma unroll
  for (int t = 0; t < 4; ++t) bB[t] = ld_b(Wr[t] + k0 + 32);
#pragma unroll
  for (int t = 0; t < 4; ++t) {
    acc[t]     = wmma_f16(a0A, bA[t], acc[t]);
    acc[4 + t] = wmma_f16(a1A, bA[t], acc[4 + t]);
  }
#pragma unroll
  for (int t = 0; t < 4; ++t) {
    acc[t]     = wmma_f16(a0B, bB[t], acc[t]);
    acc[4 + t] = wmma_f16(a1B, bB[t], acc[4 + t]);
  }

  const int rowOff = (lane >> 4) << 3;            // D layout: row = r + rowOff
#pragma unroll
  for (int mi = 0; mi < 2; ++mi)
#pragma unroll
    for (int t = 0; t < 4; ++t) {
      const int col = nBase + t * 16 + colLane;
      const float bv = bias[col];
#pragma unroll
      for (int r = 0; r < 8; ++r) {
        const float val = acc[4 * mi + t][r] + bv;
        const int row = mBase + mi * 16 + r + rowOff;
        if (mode == 0) {
          ((_Float16*)outp)[(size_t)row * CC + col] = (_Float16)val;
        } else if (mode == 1) {                   // V transposed: [B,H,HD,T]
          const int bi = row >> 11, tok = row & (TT - 1);
          const int h = col >> 6, d = col & (HD - 1);
          ((_Float16*)outp)[((size_t)(bi * HH + h) * HD + d) * TT + tok] =
              (_Float16)val;
        } else {
          ((float*)outp)[(size_t)row * CC + col] = val;
        }
      }
    }
}

// ---- Flash attention: 1 wave per (b, h, 16-query tile) ---------------------
// K operands for tile jb+32 are prefetched (loads land directly in kb[], no
// rotation copies) during the softmax VALU of tile jb; V operands are loaded
// at the top of each iteration so they are resident before the P x V WMMAs.
__global__ void __launch_bounds__(32)
attn_kernel(const _Float16* __restrict__ qh, const _Float16* __restrict__ kh,
            const _Float16* __restrict__ vt, _Float16* __restrict__ yh) {
  __shared__ _Float16 Plds[16 * 40];              // 16x32 P tile, padded stride

  const int lane    = threadIdx.x;
  const int qTile   = blockIdx.x & (TT / 16 - 1); // 128 tiles
  const int bh      = blockIdx.x >> 7;
  const int b       = bh >> 4, h = bh & 15;
  const int qBase   = qTile << 4;
  const int colLane = lane & 15;
  const int rowOff  = (lane >> 4) << 3;
  const int kA = (lane >> 4) << 3;
  const int kB = (lane >> 4) << 3;

  // Q 16x64 as two chained A operands (K = 32 each), resident in VGPRs.
  const _Float16* Qa =
      qh + ((size_t)(b * TT + qBase + colLane)) * CC + h * HD + kA;
  const v16h qa0 = ld_a(Qa);
  const v16h qa1 = ld_a(Qa + 32);

  const _Float16* Kbase = kh + ((size_t)(b * TT + colLane)) * CC + h * HD + kB;
  const _Float16* Vbase =
      vt + ((size_t)(b * HH + h) * HD + colLane) * TT + kB;

  float m_r[8], l_r[8];
  v8f acc[4] = {};
#pragma unroll
  for (int r = 0; r < 8; ++r) { m_r[r] = -1e30f; l_r[r] = 0.f; }

  const int jEnd = qBase + 16;                    // causal early exit

  // K-tile prologue (jb = 0): kb[t][chain]
  v16h kb[2][2];
#pragma unroll
  for (int t = 0; t < 2; ++t) {
    const _Float16* Kp = Kbase + (size_t)(t * 16) * CC;
    kb[t][0] = ld_b(Kp);
    kb[t][1] = ld_b(Kp + 32);
  }

  for (int jb = 0; jb < jEnd; jb += 32) {
    // ---- V operands for this tile: issue loads early ----
    v16h vb[4];
#pragma unroll
    for (int t4 = 0; t4 < 4; ++t4)
      vb[t4] = ld_b(Vbase + (size_t)(t4 * 16) * TT + jb);

    // ---- S = Q K^T : two 16x16 N-subtiles, K-chain over d=64 ----
    v8f s[2] = {};
#pragma unroll
    for (int t = 0; t < 2; ++t) {
      s[t] = wmma_f16(qa0, kb[t][0], s[t]);
      s[t] = wmma_f16(qa1, kb[t][1], s[t]);
    }

    // ---- prefetch next K tile while softmax VALU runs ----
    if (jb + 32 < jEnd) {
#pragma unroll
      for (int t = 0; t < 2; ++t) {
        const _Float16* Kp = Kbase + (size_t)(jb + 32 + t * 16) * CC;
        kb[t][0] = ld_b(Kp);
        kb[t][1] = ld_b(Kp + 32);
      }
    }

    // ---- scale + mask in D layout (lane = col j, vgpr r = row i) ----
#pragma unroll
    for (int t = 0; t < 2; ++t)
#pragma unroll
      for (int r = 0; r < 8; ++r) {
        const int j = jb + t * 16 + colLane;
        const int i = qBase + r + rowOff;
        const bool ok = (j <= i) && (j % 25 != 24);
        s[t][r] = ok ? s[t][r] * 0.125f : -1e30f;  // 1/sqrt(64)
      }
    // ---- online softmax: reductions across the 16-lane group ----
#pragma unroll
    for (int r = 0; r < 8; ++r) {
      float v = fmaxf(s[0][r], s[1][r]);
      v = fmaxf(v, __shfl_xor(v, 1));
      v = fmaxf(v, __shfl_xor(v, 2));
      v = fmaxf(v, __shfl_xor(v, 4));
      v = fmaxf(v, __shfl_xor(v, 8));
      const float mn = fmaxf(m_r[r], v);
      const float alpha = __expf(m_r[r] - mn);
      m_r[r] = mn;
      const float p0 = __expf(s[0][r] - mn);
      const float p1 = __expf(s[1][r] - mn);
      s[0][r] = p0; s[1][r] = p1;
      float rs = p0 + p1;
      rs += __shfl_xor(rs, 1);
      rs += __shfl_xor(rs, 2);
      rs += __shfl_xor(rs, 4);
      rs += __shfl_xor(rs, 8);
      l_r[r] = l_r[r] * alpha + rs;
#pragma unroll
      for (int t4 = 0; t4 < 4; ++t4) acc[t4][r] *= alpha;
    }
    // ---- D-layout -> A-layout for P via LDS ----
#pragma unroll
    for (int t = 0; t < 2; ++t)
#pragma unroll
      for (int r = 0; r < 8; ++r)
        Plds[(r + rowOff) * 40 + t * 16 + colLane] = (_Float16)s[t][r];
    __syncthreads();                               // single-wave WG: NOP + waits
    const v16h pa = ld_a(&Plds[colLane * 40 + kA]);
    // ---- acc += P @ V (V operands already resident) ----
#pragma unroll
    for (int t4 = 0; t4 < 4; ++t4)
      acc[t4] = wmma_f16(pa, vb[t4], acc[t4]);
    __syncthreads();
  }
  // ---- normalize and write y (f16, [B,T,C]) ----
#pragma unroll
  for (int t4 = 0; t4 < 4; ++t4)
#pragma unroll
    for (int r = 0; r < 8; ++r) {
      const int row = qBase + r + rowOff;
      const int col = h * HD + t4 * 16 + colLane;
      yh[((size_t)(b * TT + row)) * CC + col] = (_Float16)(acc[t4][r] / l_r[r]);
    }
}

// ---------------------------------------------------------------------------
extern "C" void kernel_launch(void* const* d_in, const int* in_sizes, int n_in,
                              void* d_out, int out_size, void* d_ws,
                              size_t ws_size, hipStream_t stream) {
  (void)in_sizes; (void)n_in; (void)out_size; (void)ws_size;
  const float* x  = (const float*)d_in[0];
  const float* Wq = (const float*)d_in[1];
  const float* bq = (const float*)d_in[2];
  const float* Wk = (const float*)d_in[3];
  const float* bk = (const float*)d_in[4];
  const float* Wv = (const float*)d_in[5];
  const float* bv = (const float*)d_in[6];
  const float* Wp = (const float*)d_in[7];
  const float* bp = (const float*)d_in[8];
  float* out = (float*)d_out;

  const size_t NXT = (size_t)BB * TT * CC;  // 8388608
  const size_t NW  = (size_t)CC * CC;       // 1048576
  _Float16* ws  = (_Float16*)d_ws;
  _Float16* xh  = ws;
  _Float16* qh  = xh  + NXT;
  _Float16* kh  = qh  + NXT;
  _Float16* vth = kh  + NXT;
  _Float16* yh  = vth + NXT;
  _Float16* wqh = yh  + NXT;
  _Float16* wkh = wqh + NW;
  _Float16* wvh = wkh + NW;
  _Float16* wph = wvh + NW;

  cvt_f32_f16<<<2048, 256, 0, stream>>>(x,  xh,  (int)(NXT / 4));
  cvt_f32_f16<<<512,  256, 0, stream>>>(Wq, wqh, (int)(NW / 4));
  cvt_f32_f16<<<512,  256, 0, stream>>>(Wk, wkh, (int)(NW / 4));
  cvt_f32_f16<<<512,  256, 0, stream>>>(Wv, wvh, (int)(NW / 4));
  cvt_f32_f16<<<512,  256, 0, stream>>>(Wp, wph, (int)(NW / 4));

  // (M/32)*(C/64) = 256*16 = 4096 waves, 4 waves/block -> 1024 blocks
  gemm_wmma<<<1024, 128, 0, stream>>>(xh, wqh, bq, qh,  0);
  gemm_wmma<<<1024, 128, 0, stream>>>(xh, wkh, bk, kh,  0);
  gemm_wmma<<<1024, 128, 0, stream>>>(xh, wvh, bv, vth, 1);

  // B*H*(T/16) = 8192 single-wave blocks
  attn_kernel<<<BB * HH * (TT / 16), 32, 0, stream>>>(qh, kh, vth, yh);

  gemm_wmma<<<1024, 128, 0, stream>>>(yh, wph, bp, out, 2);
}
